// Net_46359876993375
// MI455X (gfx1250) — matrix operand order repaired and checked by reading, not compiled, not run on hardware
//
#include <hip/hip_runtime.h>

// ---------- types ----------
typedef __attribute__((ext_vector_type(16))) __bf16 v16bf;
typedef __attribute__((ext_vector_type(8)))  float  v8f;
typedef unsigned short ushortT;

union FragBF {
    v16bf  v;
    uint4  q[2];
    ushortT us[16];
};

// ---------- scalar helpers ----------
__device__ __forceinline__ ushortT f2bf(float f) {
    unsigned u = __float_as_uint(f);
    return (ushortT)((u + 0x7FFFu + ((u >> 16) & 1u)) >> 16);   // RNE
}
__device__ __forceinline__ float b2f(ushortT h) {
    return __uint_as_float(((unsigned)h) << 16);
}
__device__ __forceinline__ int imax(int a, int b) { return a > b ? a : b; }

__device__ __forceinline__ void store_val(float* o, size_t i, float x)   { o[i] = x; }
__device__ __forceinline__ void store_val(ushortT* o, size_t i, float x) { o[i] = f2bf(x); }

// ---------- conversion kernels ----------
// W [K x N] f32  ->  Wt [N x Kpad] bf16 (transposed, zero-padded K)
__global__ void conv_w(const float* __restrict__ W, ushortT* __restrict__ Wt,
                       int K, int N, int Kpad) {
    size_t i = (size_t)blockIdx.x * blockDim.x + threadIdx.x;
    size_t total = (size_t)N * Kpad;
    if (i >= total) return;
    int n = (int)(i / Kpad), k = (int)(i % Kpad);
    Wt[i] = (k < K) ? f2bf(W[(size_t)k * N + n]) : (ushortT)0;
}

// A [M x K] f32 -> [Mpad x Kpad] bf16 row-major, zero-padded
__global__ void conv_a(const float* __restrict__ A, ushortT* __restrict__ out,
                       int M, int K, int Mpad, int Kpad) {
    size_t i = (size_t)blockIdx.x * blockDim.x + threadIdx.x;
    size_t total = (size_t)Mpad * Kpad;
    if (i >= total) return;
    int r = (int)(i / Kpad), k = (int)(i % Kpad);
    out[i] = (r < M && k < K) ? f2bf(A[(size_t)r * K + k]) : (ushortT)0;
}

// ---------- generic bf16 WMMA GEMM: out = relu(A @ W + b) [* rowMask] ----------
// A [Mpad x Kpad] bf16 row-major, Wt [N x Kpad] bf16 (W transposed).
// Block = 256 threads = 8 waves; wave w -> rows [blockIdx.x*128 + w*16, +16),
// cols [blockIdx.y*64, +64) as 4 accumulator fragments.
template <typename OutT>
__global__ void gemm_bf16(const ushortT* __restrict__ A, const ushortT* __restrict__ Wt,
                          const float* __restrict__ bias, OutT* __restrict__ out,
                          int Mpad, int N, int Kpad, const int* __restrict__ rowMask) {
    const int lane = threadIdx.x & 31;
    const int wv   = threadIdx.x >> 5;
    const int rowBase = blockIdx.x * 128 + wv * 16;
    if (rowBase >= Mpad) return;
    const int colBase = blockIdx.y * 64;
    const int r16 = lane & 15;
    const int hi  = lane >> 4;

    v8f acc[4];
#pragma unroll
    for (int f = 0; f < 4; ++f) acc[f] = (v8f){0.f,0.f,0.f,0.f,0.f,0.f,0.f,0.f};

    const ushortT* arow = A  + (size_t)(rowBase + r16) * Kpad + hi * 8;
    const ushortT* bcol = Wt + (size_t)(colBase + r16) * Kpad + hi * 8;
    const int ksteps = Kpad >> 5;

    for (int kt = 0; kt < ksteps; ++kt) {
        __builtin_prefetch(arow + kt * 32 + 64, 0, 1);
        FragBF a;
        a.q[0] = *(const uint4*)(arow + kt * 32);
        a.q[1] = *(const uint4*)(arow + kt * 32 + 16);
#pragma unroll
        for (int f = 0; f < 4; ++f) {
            const ushortT* bp = bcol + (size_t)f * 16 * Kpad + kt * 32;
            FragBF bq;
            bq.q[0] = *(const uint4*)(bp);
            bq.q[1] = *(const uint4*)(bp + 16);
            acc[f] = __builtin_amdgcn_wmma_f32_16x16x32_bf16(
                false, a.v, false, bq.v, (short)0, acc[f], false, false);
        }
    }

    const int rb = hi * 8;
#pragma unroll
    for (int f = 0; f < 4; ++f) {
        int col = colBase + f * 16 + r16;
        float bv = bias ? bias[col] : 0.f;
#pragma unroll
        for (int v = 0; v < 8; ++v) {
            int row = rowBase + rb + v;
            float x = acc[f][v] + bv;
            x = x > 0.f ? x : 0.f;                      // all first layers use relu
            if (rowMask) x *= (float)rowMask[row];
            store_val(out, (size_t)row * N + col, x);
        }
    }
}

// ---------- GEMM with gathered/fused A (per-element functor) ----------
template <typename AF, typename OutT>
__global__ void gemm_gatherA(AF af, const ushortT* __restrict__ Wt,
                             const float* __restrict__ bias, OutT* __restrict__ out,
                             int Mpad, int N, int Kpad, const int* __restrict__ rowMask) {
    const int lane = threadIdx.x & 31;
    const int wv   = threadIdx.x >> 5;
    const int rowBase = blockIdx.x * 128 + wv * 16;
    if (rowBase >= Mpad) return;
    const int colBase = blockIdx.y * 64;
    const int r16 = lane & 15;
    const int hi  = lane >> 4;
    const int rowA = rowBase + r16;

    v8f acc[4];
#pragma unroll
    for (int f = 0; f < 4; ++f) acc[f] = (v8f){0.f,0.f,0.f,0.f,0.f,0.f,0.f,0.f};

    const ushortT* bcol = Wt + (size_t)(colBase + r16) * Kpad + hi * 8;
    const int ksteps = Kpad >> 5;

    for (int kt = 0; kt < ksteps; ++kt) {
        FragBF a;
        const int kb = kt * 32 + hi * 8;
#pragma unroll
        for (int j = 0; j < 8; ++j) {
            a.us[j]     = af.get(rowA, kb + j);
            a.us[8 + j] = af.get(rowA, kb + 16 + j);
        }
#pragma unroll
        for (int f = 0; f < 4; ++f) {
            const ushortT* bp = bcol + (size_t)f * 16 * Kpad + kt * 32;
            FragBF bq;
            bq.q[0] = *(const uint4*)(bp);
            bq.q[1] = *(const uint4*)(bp + 16);
            acc[f] = __builtin_amdgcn_wmma_f32_16x16x32_bf16(
                false, a.v, false, bq.v, (short)0, acc[f], false, false);
        }
    }

    const int rb = hi * 8;
#pragma unroll
    for (int f = 0; f < 4; ++f) {
        int col = colBase + f * 16 + r16;
        float bv = bias ? bias[col] : 0.f;
#pragma unroll
        for (int v = 0; v < 8; ++v) {
            int row = rowBase + rb + v;
            float x = acc[f][v] + bv;
            x = x > 0.f ? x : 0.f;
            if (rowMask) x *= (float)rowMask[row];
            store_val(out, (size_t)row * N + col, x);
        }
    }
}

// A-fragment functors -------------------------------------------------------
// slot_in row r (258048 rows, K=258 pad 288): [slot_f(2) | move_tab[idx](256)]
struct SlotAF {
    const float* slot_f; const int* idx; const ushortT* mtab;
    __device__ __forceinline__ ushortT get(int row, int k) const {
        if (k < 2)   return f2bf(slot_f[(size_t)row * 2 + k]);
        if (k < 258) return mtab[(size_t)idx[row] * 256 + (k - 2)];
        return 0;
    }
};
// xm row q=(b*4+a)*2+t (8192 rows, K=2220 pad 2240): [battle(2091)|mopt(128)|t(1)]
struct XmAF {
    const ushortT* battle; const float* mopt;
    __device__ __forceinline__ ushortT get(int row, int k) const {
        int b = row >> 3, r = row & 7, a = r >> 1, t = r & 1;
        if (k < 2091)  return battle[(size_t)b * 2091 + k];
        if (k < 2219)  return f2bf(mopt[(size_t)(b * 4 + a) * 128 + (k - 2091)]);
        if (k == 2219) return t ? (ushortT)0x3F80 : (ushortT)0;   // bf16(1.0)
        return 0;
    }
};
// xs row q=b*6+kk (6144 rows, K=2603 pad 2624): [battle(2091)|user_x[b*12+sp](512)]
struct XsAF {
    const ushortT* battle; const ushortT* user_x; const int* sw;
    __device__ __forceinline__ ushortT get(int row, int k) const {
        int b = row / 6;
        if (k < 2091) return battle[(size_t)b * 2091 + k];
        if (k < 2603) {
            int sp = sw[b * 6 + (row % 6)];
            return user_x[(size_t)(b * 12 + sp) * 512 + (k - 2091)];
        }
        return 0;
    }
};

// ---------- pooling / user_in assembly (grid=12288 users, block=128) ----------
__global__ void user_assemble(const ushortT* __restrict__ srep,
                              const float* __restrict__ item_tab,
                              const float* __restrict__ abil_tab,
                              const float* __restrict__ user_feat,
                              const float* __restrict__ types_x,
                              const float* __restrict__ tera_x,
                              const int* __restrict__ slot_mask,
                              const int* __restrict__ item_idx,
                              const int* __restrict__ item_mask,
                              const int* __restrict__ berry_idx,
                              const int* __restrict__ abil_idx,
                              const int* __restrict__ abil_mask,
                              ushortT* __restrict__ user_in) {
    const int u = blockIdx.x;      // 0..12287  (b,s,p) flattened
    const int j = threadIdx.x;     // 0..127
    ushortT* dst = user_in + (size_t)u * 1312;
    const size_t sb = (size_t)u * 21;

    if (j < 89) dst[j] = f2bf(user_feat[(size_t)u * 89 + j]);
#pragma unroll
    for (int i = 0; i < 5; ++i)    // lookup_x = slots 16..20 (already masked)
        dst[89 + i * 128 + j] = srep[(sb + 16 + i) * 128 + j];

    dst[729 + j] = f2bf(item_tab[(size_t)berry_idx[u] * 128 + j]);   // berry_x

    // move_set_x = max(set_reps 0..3, masked-mean of slots 4..15)
    float m = -3.4e38f;
#pragma unroll
    for (int i = 0; i < 4; ++i) m = fmaxf(m, b2f(srep[(sb + i) * 128 + j]));
    float ssum = 0.f; int cnt = 0;
#pragma unroll
    for (int i = 4; i < 16; ++i) {
        ssum += b2f(srep[(sb + i) * 128 + j]);
        cnt  += slot_mask[sb + i];
    }
    m = fmaxf(m, ssum / (float)imax(cnt, 1));
    dst[857 + j] = f2bf(m);

    // item_x = masked mean of 2
    int i0 = item_idx[u * 2], i1 = item_idx[u * 2 + 1];
    int m0 = item_mask[u * 2], m1 = item_mask[u * 2 + 1];
    float it = (item_tab[(size_t)i0 * 128 + j] * m0 + item_tab[(size_t)i1 * 128 + j] * m1)
               / (float)imax(m0 + m1, 1);
    dst[985 + j] = f2bf(it);

    // abil_x = masked mean of 3
    float ab = 0.f; int am = 0;
#pragma unroll
    for (int i = 0; i < 3; ++i) {
        int ai = abil_idx[u * 3 + i], mk = abil_mask[u * 3 + i];
        ab += abil_tab[(size_t)ai * 128 + j] * mk; am += mk;
    }
    dst[1113 + j] = f2bf(ab / (float)imax(am, 1));

    if (j < 20) dst[1241 + j] = f2bf(types_x[(size_t)u * 20 + j]);
    if (j < 20) dst[1261 + j] = f2bf(tera_x[(size_t)u * 20 + j]);
    if (j < 31) dst[1281 + j] = 0;   // K padding
}

// ---------- battle_x assembly (grid=B, block=256) ----------
__global__ void battle_assemble(const float* __restrict__ obs_x,
                                const float* __restrict__ side_x,
                                const ushortT* __restrict__ user_x,
                                const int* __restrict__ active_idx,
                                ushortT* __restrict__ battle) {
    int b = blockIdx.x, t = threadIdx.x;
    ushortT* dst = battle + (size_t)b * 2091;
    if (t < 9) dst[t] = f2bf(obs_x[b * 9 + t]);
#pragma unroll
    for (int s = 0; s < 2; ++s) {
        int base = 9 + s * 1041;
        if (t < 17) dst[base + t] = f2bf(side_x[(b * 2 + s) * 17 + t]);
        int ai = active_idx[b * 2 + s];
        for (int j = t; j < 512; j += 256) {
            dst[base + 17 + j] = user_x[(size_t)(b * 12 + s * 6 + ai) * 512 + j];
            float mx = -3.4e38f;
#pragma unroll
            for (int p = 0; p < 6; ++p)
                mx = fmaxf(mx, b2f(user_x[(size_t)(b * 12 + s * 6 + p) * 512 + j]));
            dst[base + 17 + 512 + j] = f2bf(mx);
        }
    }
}

// ---------- head layer-2 dots + masking -> d_out (one wave per logit) ----------
__global__ void head_logits(const ushortT* __restrict__ hm, const ushortT* __restrict__ hs,
                            const float* __restrict__ Wm2, const float* __restrict__ bm2,
                            const float* __restrict__ Ws2, const float* __restrict__ bs2,
                            const int* __restrict__ move_valid,
                            const int* __restrict__ switch_valid,
                            const int* __restrict__ tera_flag,
                            float* __restrict__ out) {
    int gtid = blockIdx.x * blockDim.x + threadIdx.x;
    int w = gtid >> 5, lane = gtid & 31;
    if (w >= 8192 + 6144) return;
    float s = 0.f;
    if (w < 8192) {
        const ushortT* row = hm + (size_t)w * 512;
        for (int k = lane; k < 512; k += 32) s += b2f(row[k]) * Wm2[k];
        for (int off = 16; off > 0; off >>= 1) s += __shfl_down(s, off, 32);
        if (lane == 0) {
            int b = w >> 3, r = w & 7, a = r >> 1, t = r & 1;
            bool valid = (move_valid[b * 4 + a] > 0) && ((tera_flag[b] > 0) || (t == 0));
            out[b * 14 + r] = valid ? (s + bm2[0]) : -__builtin_inff();
        }
    } else {
        int q = w - 8192;
        const ushortT* row = hs + (size_t)q * 512;
        for (int k = lane; k < 512; k += 32) s += b2f(row[k]) * Ws2[k];
        for (int off = 16; off > 0; off >>= 1) s += __shfl_down(s, off, 32);
        if (lane == 0) {
            int b = q / 6, kk = q % 6;
            out[b * 14 + 8 + kk] = (switch_valid[b * 6 + kk] > 0) ? (s + bs2[0])
                                                                  : -__builtin_inff();
        }
    }
}

// ---------- host launcher ----------
extern "C" void kernel_launch(void* const* d_in, const int* in_sizes, int n_in,
                              void* d_out, int out_size, void* d_ws, size_t ws_size,
                              hipStream_t stream) {
    (void)in_sizes; (void)n_in; (void)out_size; (void)ws_size;
    const float* obs_x      = (const float*)d_in[0];
    const float* side_x     = (const float*)d_in[1];
    const float* user_feat  = (const float*)d_in[2];
    const float* slot_f     = (const float*)d_in[3];
    const float* types_x    = (const float*)d_in[4];
    const float* tera_x     = (const float*)d_in[5];
    const float* mopt       = (const float*)d_in[6];
    const float* dex_moves  = (const float*)d_in[7];
    const float* dex_items  = (const float*)d_in[8];
    const float* dex_abil   = (const float*)d_in[9];
    const float* W_item = (const float*)d_in[10]; const float* b_item = (const float*)d_in[11];
    const float* W_abil = (const float*)d_in[12]; const float* b_abil = (const float*)d_in[13];
    const float* W_move = (const float*)d_in[14]; const float* b_move = (const float*)d_in[15];
    const float* W_slot = (const float*)d_in[16]; const float* b_slot = (const float*)d_in[17];
    const float* W_user = (const float*)d_in[18]; const float* b_user = (const float*)d_in[19];
    const float* W_m1   = (const float*)d_in[20]; const float* b_m1   = (const float*)d_in[21];
    const float* W_m2   = (const float*)d_in[22]; const float* b_m2   = (const float*)d_in[23];
    const float* W_s1   = (const float*)d_in[24]; const float* b_s1   = (const float*)d_in[25];
    const float* W_s2   = (const float*)d_in[26]; const float* b_s2   = (const float*)d_in[27];
    const int* slot_move_idx = (const int*)d_in[28];
    const int* slot_mask     = (const int*)d_in[29];
    const int* item_idx      = (const int*)d_in[30];
    const int* item_mask     = (const int*)d_in[31];
    const int* berry_idx     = (const int*)d_in[32];
    const int* abil_idx      = (const int*)d_in[33];
    const int* abil_mask     = (const int*)d_in[34];
    const int* active_idx    = (const int*)d_in[35];
    const int* switch_sp     = (const int*)d_in[36];
    const int* move_valid    = (const int*)d_in[37];
    const int* switch_valid  = (const int*)d_in[38];
    const int* tera_flag     = (const int*)d_in[39];
    float* out = (float*)d_out;

    // bump allocator over workspace (~140 MB total)
    char* p = (char*)d_ws;
    auto alloc = [&](size_t bytes) -> void* {
        void* r = (void*)p;
        p += (bytes + 255) & ~(size_t)255;
        return r;
    };
    ushortT* Wt_move = (ushortT*)alloc((size_t)256 * 544 * 2);
    ushortT* Wt_slot = (ushortT*)alloc((size_t)128 * 288 * 2);
    ushortT* Wt_item = (ushortT*)alloc((size_t)128 * 256 * 2);
    ushortT* Wt_abil = (ushortT*)alloc((size_t)128 * 256 * 2);
    ushortT* Wt_user = (ushortT*)alloc((size_t)512 * 1312 * 2);
    ushortT* Wt_m1   = (ushortT*)alloc((size_t)512 * 2240 * 2);
    ushortT* Wt_s1   = (ushortT*)alloc((size_t)512 * 2624 * 2);
    ushortT* A_moves = (ushortT*)alloc((size_t)912 * 544 * 2);
    ushortT* A_items = (ushortT*)alloc((size_t)304 * 256 * 2);
    ushortT* A_abil  = (ushortT*)alloc((size_t)304 * 256 * 2);
    ushortT* move_tab = (ushortT*)alloc((size_t)912 * 256 * 2);
    float*   item_tab = (float*)alloc((size_t)304 * 128 * 4);
    float*   abil_tab = (float*)alloc((size_t)304 * 128 * 4);
    ushortT* srep    = (ushortT*)alloc((size_t)258048 * 128 * 2);
    ushortT* user_in = (ushortT*)alloc((size_t)12288 * 1312 * 2);
    ushortT* user_x  = (ushortT*)alloc((size_t)12288 * 512 * 2);
    ushortT* battle  = (ushortT*)alloc((size_t)1024 * 2091 * 2);
    ushortT* hm      = (ushortT*)alloc((size_t)8192 * 512 * 2);
    ushortT* hs      = (ushortT*)alloc((size_t)6144 * 512 * 2);

    auto g1 = [](size_t n) { return dim3((unsigned)((n + 255) / 256)); };

    // weight transposes + padding
    conv_w<<<g1((size_t)256 * 544), 256, 0, stream>>>(W_move, Wt_move, 514, 256, 544);
    conv_w<<<g1((size_t)128 * 288), 256, 0, stream>>>(W_slot, Wt_slot, 258, 128, 288);
    conv_w<<<g1((size_t)128 * 256), 256, 0, stream>>>(W_item, Wt_item, 256, 128, 256);
    conv_w<<<g1((size_t)128 * 256), 256, 0, stream>>>(W_abil, Wt_abil, 256, 128, 256);
    conv_w<<<g1((size_t)512 * 1312), 256, 0, stream>>>(W_user, Wt_user, 1281, 512, 1312);
    conv_w<<<g1((size_t)512 * 2240), 256, 0, stream>>>(W_m1, Wt_m1, 2220, 512, 2240);
    conv_w<<<g1((size_t)512 * 2624), 256, 0, stream>>>(W_s1, Wt_s1, 2603, 512, 2624);
    // dex tables -> bf16 padded A
    conv_a<<<g1((size_t)912 * 544), 256, 0, stream>>>(dex_moves, A_moves, 900, 514, 912, 544);
    conv_a<<<g1((size_t)304 * 256), 256, 0, stream>>>(dex_items, A_items, 300, 256, 304, 256);
    conv_a<<<g1((size_t)304 * 256), 256, 0, stream>>>(dex_abil,  A_abil,  300, 256, 304, 256);

    // table GEMMs (gather commutes with row-matmul)
    gemm_bf16<ushortT><<<dim3(8, 4), 256, 0, stream>>>(A_moves, Wt_move, b_move, move_tab, 912, 256, 544, nullptr);
    gemm_bf16<float>  <<<dim3(3, 2), 256, 0, stream>>>(A_items, Wt_item, b_item, item_tab, 304, 128, 256, nullptr);
    gemm_bf16<float>  <<<dim3(3, 2), 256, 0, stream>>>(A_abil,  Wt_abil, b_abil, abil_tab, 304, 128, 256, nullptr);

    // srep: fused slot_in gather, relu, row mask  [258048 x 128]
    SlotAF saf{slot_f, slot_move_idx, move_tab};
    gemm_gatherA<SlotAF, ushortT><<<dim3(2016, 2), 256, 0, stream>>>(saf, Wt_slot, b_slot, srep, 258048, 128, 288, slot_mask);

    // user_in assembly (pool/mean/max/gathers)
    user_assemble<<<12288, 128, 0, stream>>>(srep, item_tab, abil_tab, user_feat, types_x, tera_x,
                                             slot_mask, item_idx, item_mask, berry_idx,
                                             abil_idx, abil_mask, user_in);

    // user_x = relu(user_in @ W_user + b)  [12288 x 512]
    gemm_bf16<ushortT><<<dim3(96, 8), 256, 0, stream>>>(user_in, Wt_user, b_user, user_x, 12288, 512, 1312, nullptr);

    // battle_x [B x 2091] bf16
    battle_assemble<<<1024, 256, 0, stream>>>(obs_x, side_x, user_x, active_idx, battle);

    // heads layer 1 with fused xm / xs gathers
    XmAF xmf{battle, mopt};
    gemm_gatherA<XmAF, ushortT><<<dim3(64, 8), 256, 0, stream>>>(xmf, Wt_m1, b_m1, hm, 8192, 512, 2240, nullptr);
    XsAF xsf{battle, user_x, switch_sp};
    gemm_gatherA<XsAF, ushortT><<<dim3(48, 8), 256, 0, stream>>>(xsf, Wt_s1, b_s1, hs, 6144, 512, 2624, nullptr);

    // layer-2 dots + masking -> logits [B x 14]
    head_logits<<<1792, 256, 0, stream>>>(hm, hs, W_m2, b_m2, W_s2, b_s2,
                                          move_valid, switch_valid, tera_flag, out);
}
